// LinearAttention_69140383531108
// MI455X (gfx1250) — matrix operand order repaired and checked by reading, not compiled, not run on hardware
//
#include <hip/hip_runtime.h>
#include <hip/hip_bf16.h>

// ---------------------------------------------------------------------------
// DeltaNet-style chunked linear attention for MI455X (gfx1250, wave32, WMMA).
//   B*H = 32 heads -> one workgroup (256 thr = 8 waves) per head.
//   Per chunk: A = tril(we ke^T,-1) [WMMA bf16], blocked forward substitution
//   (I+A)X=[we|ue] [WMMA trailing updates + register in-block solve],
//   3x (64x64x64) state matmuls [WMMA], elementwise finalize, rank-2 state
//   update + unlinear gate.
// All WMMA operands live in LDS as bf16, laid out so each lane's fragment is
// contiguous: A-operand = 2x ds_load_b128 from row-major bf16; B-operand =
// 2x ds_load_b128 from an N-major (transposed) bf16 copy. f32 master copies
// kept only where precision matters (solve RHS, elementwise epilogue).
// wid is readfirstlane'd so all WMMA control flow is scalar (EXEC all-ones).
// ~277 KB dynamic LDS (<= 320 KB/WGP).
// ---------------------------------------------------------------------------

typedef __attribute__((ext_vector_type(16))) __bf16 v16bf;
typedef __attribute__((ext_vector_type(2)))  __bf16 bf2;
typedef __attribute__((ext_vector_type(8)))  float  v8f;

#define LDA 132   // f32 stride for 128-wide rows
#define LDB 68    // f32 stride for 64-wide rows
#define LDT 136   // bf16 stride for 128-wide rows (8-elem aligned)
#define LDW 72    // bf16 stride for 64-wide rows  (8-elem aligned)
#define SMEM_BYTES 283392

union V16U { uint4 q2[2]; v16bf v; };

// A operand 16x32 bf16 (code's own contiguous k order, matched by B; not the ISA table): lane l (m=l&15, hi=l>>4), v16bf elem
// i maps to K = 16*hi+i. Two contiguous 8xbf16 chunks.
// Source row index = rmul*(M0+m)+radd (rmul=2, radd=0/1 -> even/odd rows).
__device__ __forceinline__ v16bf pack_a_bf(const __bf16* base, int ld, int rmul,
                                           int radd, int M0, int K0, int kmax) {
  int l = threadIdx.x & 31;
  int m = l & 15, hi = l >> 4;
  const __bf16* row = base + (rmul * (M0 + m) + radd) * ld + K0;
  V16U u;
  if (16 * hi < kmax) { u.q2[0] = *(const uint4*)(row + 16 * hi);      // K = K0+16hi .. +15
                        u.q2[1] = *(const uint4*)(row + 16 * hi + 8); }
  else { u.q2[0] = make_uint4(0, 0, 0, 0); u.q2[1] = make_uint4(0, 0, 0, 0); }  // zero-pad K >= kmax
  return u.v;
}

// B operand 32x16 bf16: lane l (n=l&15, hi=l>>4), elem i -> K = K0+16*hi+i.
// Source stored N-major: base[n*ld + k] = B[k][n]. Two contiguous chunks.
__device__ __forceinline__ v16bf pack_b_bf(const __bf16* base, int ld, int K0,
                                           int N0, int kmax) {
  int l = threadIdx.x & 31;
  int n = l & 15, hi = l >> 4;
  V16U u;
  if (16 * hi < kmax) {
    const __bf16* row = base + (N0 + n) * ld + K0 + 16 * hi;
    u.q2[0] = *(const uint4*)(row);
    u.q2[1] = *(const uint4*)(row + 8);
  } else {
    u.q2[0] = make_uint4(0, 0, 0, 0);
    u.q2[1] = make_uint4(0, 0, 0, 0);
  }
  return u.v;
}

__device__ __forceinline__ v8f wmma_bf16(v16bf a, v16bf b, v8f c) {
  return __builtin_amdgcn_wmma_f32_16x16x32_bf16(false, a, false, b,
                                                 (short)0, c, false, false);
}

__device__ __forceinline__ void st_bf2(__bf16* p, float a, float b) {
  bf2 v; v.x = (__bf16)a; v.y = (__bf16)b;     // -> v_cvt_pk_bf16_f32 + b32 st
  *(bf2*)p = v;
}

__device__ __forceinline__ float stab1(float x) {
  if (!(x == x)) x = 0.0f;                      // nan -> 0
  return fminf(fmaxf(x, -10000.0f), 10000.0f);  // clip +-1e4 (handles inf)
}

__device__ __forceinline__ float gelu_gate(float x, float xn) {
  // _unlinear: x/2 * gelu_tanh(2x/xn)
  float g = 2.0f * x / xn;
  float gl = 0.5f * g *
             (1.0f + tanhf(0.7978845608028654f * (g + 0.044715f * g * g * g)));
  return 0.5f * x * gl;
}

__global__ void __launch_bounds__(256)
delta2_kernel(const float* __restrict__ Qg, const float* __restrict__ Kg,
              const float* __restrict__ Vg, const float* __restrict__ Bg,
              float* __restrict__ Og, int S) {
  extern __shared__ char smemraw[];
  float*  WU   = (float*)smemraw;              // 128xLDA f32: [we|ue]->[wci|uci]
  __bf16* ALbf = (__bf16*)(WU + 128 * LDA);    // 128xLDT: strict-lower A (bf16)
  __bf16* WUT  = ALbf + 128 * LDT;             // 128xLDT: X^T bf16 (B operand)
  __bf16* WUbf = WUT  + 128 * LDT;             // 128xLDW: we -> wci bf16 (A op)
  __bf16* KEbf = WUbf + 128 * LDW;             // 128xLDW: ke bf16, N-major use
  __bf16* QObf = KEbf + 128 * LDW;             //  64xLDW: q_odd bf16 (A op)
  __bf16* HTbf = QObf + 64 * LDW;              //  64xLDW: h^T bf16 (B operand)
  float*  QO   = (float*)(HTbf + 64 * LDW);    //  64xLDB: q_odd f32
  float*  H    = QO  + 64 * LDB;               //  64xLDB: carried state h
  float*  U0   = H   + 64 * LDB;               //  64xLDB: KH -> uval0
  float*  UV1  = U0  + 64 * LDB;               //  64xLDB: W1H -> uval1
  float*  QH   = UV1 + 64 * LDB;               //  64xLDB: q_odd @ h
  float*  APART = QH + 64 * LDB;               // 64*4
  float*  BPART = APART + 256;                 // 64*4
  float*  AV  = BPART + 256;                   // 64
  float*  BV  = AV + 64;                       // 64
  float*  NP  = BV + 64;                       // 64*4
  float*  XN  = NP + 256;                      // 64

  const int tid  = threadIdx.x;
  const int lane = tid & 31;
  // Wave index as an explicit scalar: keeps all WMMA control flow on the
  // SALU path and EXEC at all-ones inside WMMA regions (ISA 7.12 req).
  const int wid  = __builtin_amdgcn_readfirstlane(tid >> 5);
  const int head = blockIdx.x;
  const long hq = (long)head * S * 64;
  const long hb = (long)head * S;

  for (int i = tid; i < 64 * 64; i += 256) {
    int d = i >> 6, e = i & 63;
    H[d * LDB + e] = 0.0f;
    HTbf[d * LDW + e] = (__bf16)0.0f;
  }
  __syncthreads();

  const int NCH = S / 64;
  for (int ci = 0; ci < NCH; ++ci) {
    const int t0 = ci * 64;

    // ---------------- Phase L: stage & expand chunk into LDS --------------
    // virtual row 2c = -0.5*x_t (t=t0+c); row 2c+1 = +0.5*x_{t-1}; beta_t.
    for (int i = tid; i < 64 * 32; i += 256) {
      int c = i >> 5, dp = (i & 31) << 1;       // element pairs (float2/bf2)
      int t = t0 + c, tm = t - 1;
      float bet = Bg[hb + t];
      float2 kc = *(const float2*)&Kg[hq + (long)t * 64 + dp];
      float2 vc = *(const float2*)&Vg[hq + (long)t * 64 + dp];
      float2 kp = make_float2(0.f, 0.f), vp = kp, qp = kp;
      if (tm >= 0) {
        kp = *(const float2*)&Kg[hq + (long)tm * 64 + dp];
        vp = *(const float2*)&Vg[hq + (long)tm * 64 + dp];
        qp = *(const float2*)&Qg[hq + (long)tm * 64 + dp];
      }
      float w0x = -0.5f * kc.x * bet, w0y = -0.5f * kc.y * bet;
      float w1x =  0.5f * kp.x * bet, w1y =  0.5f * kp.y * bet;
      *(float2*)&WU[(2 * c) * LDA + dp]          = make_float2(w0x, w0y);
      *(float2*)&WU[(2 * c) * LDA + 64 + dp]     = make_float2(-0.5f * vc.x * bet, -0.5f * vc.y * bet);
      *(float2*)&WU[(2 * c + 1) * LDA + dp]      = make_float2(w1x, w1y);
      *(float2*)&WU[(2 * c + 1) * LDA + 64 + dp] = make_float2(0.5f * vp.x * bet, 0.5f * vp.y * bet);
      st_bf2(&WUbf[(2 * c) * LDW + dp],     w0x, w0y);
      st_bf2(&WUbf[(2 * c + 1) * LDW + dp], w1x, w1y);
      st_bf2(&KEbf[(2 * c) * LDW + dp],     -0.5f * kc.x, -0.5f * kc.y);
      st_bf2(&KEbf[(2 * c + 1) * LDW + dp],  0.5f * kp.x,  0.5f * kp.y);
      *(float2*)&QO[c * LDB + dp] = make_float2(0.5f * qp.x, 0.5f * qp.y);
      st_bf2(&QObf[c * LDW + dp], 0.5f * qp.x, 0.5f * qp.y);
      if (ci + 1 < NCH) {   // pull next chunk toward GL2 (global_prefetch_b8)
        __builtin_prefetch(&Kg[hq + (long)(t + 64) * 64 + dp], 0, 1);
        __builtin_prefetch(&Vg[hq + (long)(t + 64) * 64 + dp], 0, 1);
      }
    }
    __syncthreads();

    // ---------------- Phase A: A = strict_lower(we @ ke^T) ----------------
    // A op: WUbf rows (we); B op: KEbf rows are ke rows = (ke^T) N-major.
    {
      int tcount = 0;
      for (int tM = 0; tM < 8; ++tM)
        for (int tN = 0; tN <= tM; ++tN) {
          if ((tcount++ & 7) != wid) continue;   // scalar branch (wid in SGPR)
          v8f acc = {0.f,0.f,0.f,0.f,0.f,0.f,0.f,0.f};
          acc = wmma_bf16(pack_a_bf(WUbf, LDW, 1, 0, tM * 16, 0, 32),
                          pack_b_bf(KEbf, LDW, 0, tN * 16, 32), acc);
          acc = wmma_bf16(pack_a_bf(WUbf, LDW, 1, 0, tM * 16, 32, 32),
                          pack_b_bf(KEbf, LDW, 32, tN * 16, 32), acc);
          int n = lane & 15, hi = lane >> 4;
#pragma unroll
          for (int r = 0; r < 8; ++r) {
            int gr = tM * 16 + r + 8 * hi, gc = tN * 16 + n;
            ALbf[gr * LDT + gc] = (__bf16)((gr > gc) ? acc[r] : 0.0f);
          }
        }
    }
    __syncthreads();

    // ------- Phase S: blocked forward substitution (I+A)X = [we|ue] -------
    for (int b = 0; b < 8; ++b) {
      if (b > 0) {
        // X_b -= A[b,<b] @ X_{<b}; wave wid owns 16 columns of the RHS
        v8f acc = {0.f,0.f,0.f,0.f,0.f,0.f,0.f,0.f};
        int Ktot = 16 * b;
        for (int kk = 0; kk < Ktot; kk += 32) {
          int kmax = Ktot - kk; if (kmax > 32) kmax = 32;
          acc = wmma_bf16(pack_a_bf(ALbf, LDT, 1, 0, b * 16, kk, kmax),
                          pack_b_bf(WUT, LDT, kk, wid * 16, kmax), acc);
        }
        int n = lane & 15, hi = lane >> 4;
#pragma unroll
        for (int r = 0; r < 8; ++r)
          WU[(b * 16 + r + 8 * hi) * LDA + wid * 16 + n] -= acc[r];
      }
      __syncthreads();
      if (wid == 0) {  // in-block 16-row serial solve; lane owns 4 columns
        float x[16][4];
        int R0 = b * 16;
#pragma unroll
        for (int r = 0; r < 16; ++r)
          *(float4*)&x[r][0] = *(const float4*)&WU[(R0 + r) * LDA + lane * 4];
        for (int r = 1; r < 16; ++r)
          for (int j = 0; j < r; ++j) {
            float aij = (float)ALbf[(R0 + r) * LDT + R0 + j];
#pragma unroll
            for (int c4 = 0; c4 < 4; ++c4) x[r][c4] -= aij * x[j][c4];
          }
#pragma unroll
        for (int r = 0; r < 16; ++r) {
          if (r) *(float4*)&WU[(R0 + r) * LDA + lane * 4] = *(float4*)&x[r][0];
#pragma unroll
          for (int c4 = 0; c4 < 4; ++c4)   // transposed bf16 copy (B operand)
            WUT[(lane * 4 + c4) * LDT + R0 + r] = (__bf16)x[r][c4];
        }
        if (lane < 16)                      // wci bf16 row-major (A operand)
#pragma unroll
          for (int r = 0; r < 16; ++r)
#pragma unroll
            for (int c4 = 0; c4 < 4; ++c4)
              WUbf[(R0 + r) * LDW + lane * 4 + c4] = (__bf16)x[r][c4];
      }
      __syncthreads();
    }
    // WU holds [wci|uci] (f32); WUbf = bf16 wci; WUT = bf16 X^T.

    // ------ Phase O: KH = wci_e@h -> U0, W1H = wci_o@h -> UV1, QH = q1@h ---
    for (int job = wid; job < 48; job += 8) {   // scalar loop (wid in SGPR)
      int which = job >> 4;                     // 0:KH 1:W1H 2:QH
      int t = job & 15;
      int tM = t >> 2, tN = t & 3;
      v8f acc = {0.f,0.f,0.f,0.f,0.f,0.f,0.f,0.f};
#pragma unroll
      for (int ks = 0; ks < 2; ++ks) {
        v16bf a;
        if (which == 0)      a = pack_a_bf(WUbf, LDW, 2, 0, tM * 16, ks * 32, 32);
        else if (which == 1) a = pack_a_bf(WUbf, LDW, 2, 1, tM * 16, ks * 32, 32);
        else                 a = pack_a_bf(QObf, LDW, 1, 0, tM * 16, ks * 32, 32);
        acc = wmma_bf16(a, pack_b_bf(HTbf, LDW, ks * 32, tN * 16, 32), acc);
      }
      float* dst = (which == 0) ? U0 : (which == 1) ? UV1 : QH;
      int n = lane & 15, hi = lane >> 4;
#pragma unroll
      for (int r = 0; r < 8; ++r)
        dst[(tM * 16 + r + 8 * hi) * LDB + tN * 16 + n] = acc[r];
    }

    // ---------------- Phase E1: row dots a_c = w1.w0, b_c = q1.w0 ---------
    {
      int c = tid >> 2, qp = tid & 3;
      const float* w0 = &WU[(2 * c) * LDA];
      const float* w1 = &WU[(2 * c + 1) * LDA];
      const float* q1 = &QO[c * LDB];
      float sa = 0.0f, sb = 0.0f;
#pragma unroll
      for (int d = 0; d < 16; ++d) {
        int dd = qp * 16 + d;
        sa += w1[dd] * w0[dd];
        sb += q1[dd] * w0[dd];
      }
      APART[c * 4 + qp] = sa;
      BPART[c * 4 + qp] = sb;
    }
    __syncthreads();
    if (tid < 64) {
      AV[tid] = APART[tid*4] + APART[tid*4+1] + APART[tid*4+2] + APART[tid*4+3];
      BV[tid] = BPART[tid*4] + BPART[tid*4+1] + BPART[tid*4+2] + BPART[tid*4+3];
    }
    __syncthreads();

    // ------------- Phase E2: finalize u0/uv1, emit outputs (float2) -------
    for (int i = tid; i < 64 * 32; i += 256) {
      int c = i >> 5, ep = (i & 31) << 1;
      float2 uce = *(const float2*)&WU[(2 * c) * LDA + 64 + ep];      // uci_e
      float2 uco = *(const float2*)&WU[(2 * c + 1) * LDA + 64 + ep];  // uci_o
      float2 kh  = *(const float2*)&U0[c * LDB + ep];                 // KH
      float2 w1h = *(const float2*)&UV1[c * LDB + ep];                // W1H
      float2 qh  = *(const float2*)&QH[c * LDB + ep];
      float2 q1  = *(const float2*)&QO[c * LDB + ep];
      float av = AV[c], bv = BV[c];
      float u0x = uce.x - kh.x, u0y = uce.y - kh.y;
      float v1x = uco.x - w1h.x - av * u0x;
      float v1y = uco.y - w1h.y - av * u0y;
      *(float2*)&U0[c * LDB + ep]  = make_float2(u0x, u0y);
      *(float2*)&UV1[c * LDB + ep] = make_float2(v1x, v1y);
      *(float2*)&Og[hq + (long)(t0 + c) * 64 + ep] =
          make_float2(qh.x + bv * u0x + q1.x * v1x,
                      qh.y + bv * u0y + q1.y * v1y);
    }
    __syncthreads();

    // ------- Phase H: rank-2 state update at c=63, stab, unlinear ---------
    for (int i = tid; i < 64 * 32; i += 256) {
      int d = i >> 5, ep = (i & 31) << 1;
      float w0 = WU[126 * LDA + d], w1 = WU[127 * LDA + d];
      float2 u0 = *(const float2*)&U0[63 * LDB + ep];
      float2 v1 = *(const float2*)&UV1[63 * LDB + ep];
      float2 hh = *(const float2*)&H[d * LDB + ep];
      float hx = stab1(hh.x + w0 * u0.x) + w1 * v1.x;
      float hy = stab1(hh.y + w0 * u0.y) + w1 * v1.y;
      *(float2*)&H[d * LDB + ep] = make_float2(stab1(hx), stab1(hy));
    }
    __syncthreads();
    {
      int d = tid >> 2, qp = tid & 3;
      float s = 0.0f;
#pragma unroll
      for (int j = 0; j < 16; ++j) {
        float x = H[d * LDB + qp * 16 + j];
        s += x * x;
      }
      NP[d * 4 + qp] = s;
    }
    __syncthreads();
    if (tid < 64)
      XN[tid] = sqrtf(NP[tid*4] + NP[tid*4+1] + NP[tid*4+2] + NP[tid*4+3]) + 1e-6f;
    __syncthreads();
    for (int i = tid; i < 64 * 32; i += 256) {
      int d = i >> 5, ep = (i & 31) << 1;
      float xn = XN[d];
      float2 hh = *(const float2*)&H[d * LDB + ep];
      float hx = gelu_gate(hh.x, xn);
      float hy = gelu_gate(hh.y, xn);
      *(float2*)&H[d * LDB + ep] = make_float2(hx, hy);
      HTbf[ep * LDW + d]       = (__bf16)hx;   // transposed bf16 (next B op)
      HTbf[(ep + 1) * LDW + d] = (__bf16)hy;
    }
    __syncthreads();
  }
}

extern "C" void kernel_launch(void* const* d_in, const int* in_sizes, int n_in,
                              void* d_out, int out_size, void* d_ws, size_t ws_size,
                              hipStream_t stream) {
  const float* q    = (const float*)d_in[0];
  const float* k    = (const float*)d_in[1];
  const float* v    = (const float*)d_in[2];
  const float* beta = (const float*)d_in[3];
  float* out = (float*)d_out;

  const int S = 2048;                       // fixed by reference
  int heads = in_sizes[3] / S;              // B*H = 32
  if (heads <= 0) return;

  hipFuncSetAttribute((const void*)delta2_kernel,
                      hipFuncAttributeMaxDynamicSharedMemorySize, SMEM_BYTES);
  delta2_kernel<<<dim3(heads), dim3(256), SMEM_BYTES, stream>>>(q, k, v, beta,
                                                                out, S);
}
